// SupConLoss_19361712571094
// MI455X (gfx1250) — matrix operand order repaired and checked by reading, not compiled, not run on hardware
//
#include <hip/hip_runtime.h>
#include <hip/hip_bf16.h>

typedef _Float16 v4h  __attribute__((ext_vector_type(4)));
typedef _Float16 v8h  __attribute__((ext_vector_type(8)));
typedef _Float16 v16h __attribute__((ext_vector_type(16)));
typedef float    v8f  __attribute__((ext_vector_type(8)));

#define DIM   128
#define KPOS  4
#define NNEG  20
#define NSLOT 24     // valid slots per row
#define TEMP_INV 10.0f

// ---------------------------------------------------------------------------
// Zero the (single-float) output; harness poisons d_out before timing.
// ---------------------------------------------------------------------------
__global__ void zero_out_kernel(float* out, int n) {
    int i = blockIdx.x * blockDim.x + threadIdx.x;
    if (i < n) out[i] = 0.0f;
}

// ---------------------------------------------------------------------------
// Row-wise L2 normalize (f32 -> f16). One wave (32 lanes) per row of 128.
// Each lane owns 4 contiguous floats; wave shuffle reduction for sum(x^2).
// ---------------------------------------------------------------------------
__global__ void normalize_rows_f16(const float* __restrict__ in,
                                   _Float16* __restrict__ out, int nrows) {
    int wave = (blockIdx.x * blockDim.x + threadIdx.x) >> 5;
    int lane = threadIdx.x & 31;
    if (wave >= nrows) return;  // uniform per wave

    const float4 v = reinterpret_cast<const float4*>(in + (size_t)wave * DIM)[lane];
    float s = v.x * v.x + v.y * v.y + v.z * v.z + v.w * v.w;
    #pragma unroll
    for (int off = 16; off > 0; off >>= 1)
        s += __shfl_xor(s, off, 32);
    float r = rsqrtf(s);

    v4h o;
    o[0] = (_Float16)(v.x * r);
    o[1] = (_Float16)(v.y * r);
    o[2] = (_Float16)(v.z * r);
    o[3] = (_Float16)(v.w * r);
    reinterpret_cast<v4h*>(out + (size_t)wave * DIM)[lane] = o;
}

// ---------------------------------------------------------------------------
// Per row: find the 4 positive class ids (ascending, matching top_k on a
// multi-hot row), then sample 20 distinct negatives (hash RNG, rejection).
// sel layout: [0..3]=positives, [4..23]=negatives, [24..31]=padding.
// labels scan is the HBM-roofline term: 268 MB total.
// ---------------------------------------------------------------------------
__global__ void select_indices_kernel(const int* __restrict__ labels,
                                      int* __restrict__ sel, int C) {
    __shared__ int cnt;
    __shared__ int pbuf[16];
    int b = blockIdx.x;
    if (threadIdx.x == 0) cnt = 0;
    __syncthreads();

    const int4* row4 = reinterpret_cast<const int4*>(labels + (size_t)b * C);
    int n4 = C >> 2;
    for (int i = threadIdx.x; i < n4; i += blockDim.x) {
        int4 l = row4[i];
        int base = i << 2;
        if (l.x) { int k = atomicAdd(&cnt, 1); if (k < 16) pbuf[k] = base + 0; }
        if (l.y) { int k = atomicAdd(&cnt, 1); if (k < 16) pbuf[k] = base + 1; }
        if (l.z) { int k = atomicAdd(&cnt, 1); if (k < 16) pbuf[k] = base + 2; }
        if (l.w) { int k = atomicAdd(&cnt, 1); if (k < 16) pbuf[k] = base + 3; }
    }
    __syncthreads();

    if (threadIdx.x == 0) {
        int tmp[32];
        int n = cnt < KPOS ? cnt : KPOS;
        for (int i = 0; i < n; i++) tmp[i] = pbuf[i];
        for (int i = n; i < KPOS; i++) tmp[i] = i;      // degenerate fallback
        // ascending sort of the 4 positives (matches top_k tie order)
        for (int i = 1; i < KPOS; i++) {
            int key = tmp[i], j = i - 1;
            while (j >= 0 && tmp[j] > key) { tmp[j + 1] = tmp[j]; j--; }
            tmp[j + 1] = key;
        }
        // distinct uniform negatives via hash + rejection
        unsigned st = (unsigned)b * 2654435761u ^ 0x9e3779b9u;
        int nsel = KPOS, tries = 0;
        while (nsel < NSLOT && tries < 4096) {
            st = st * 1664525u + 1013904223u;
            tries++;
            unsigned h = st ^ (st >> 16);
            h *= 0x7feb352du; h ^= h >> 15;
            int cand = (int)(h % (unsigned)C);
            bool dup = false;
            for (int i = 0; i < nsel; i++) if (tmp[i] == cand) { dup = true; break; }
            if (!dup) tmp[nsel++] = cand;
        }
        for (int i = nsel; i < 32; i++) tmp[i] = tmp[0];  // padding slots
        int* srow = sel + b * 32;
        for (int i = 0; i < 32; i++) srow[i] = tmp[i];
    }
}

// ---------------------------------------------------------------------------
// stable BCE-with-logits term
// ---------------------------------------------------------------------------
__device__ __forceinline__ float bce_term(float z, float t) {
    return fmaxf(z, 0.0f) - z * t + log1pf(expf(-fabsf(z)));
}

// ---------------------------------------------------------------------------
// Gathered mini-GEMM on WMMA. One wave per feature row.
// A (16x32 f16): rows = 16 selected prototypes (gathered); B (32x16 f16):
// every column = f_b. D = A*B -> column 0 (lanes 0 & 16) holds the 16 dots.
// Two slot groups x 4 K-steps = 8 v_wmma_f32_16x16x32_f16 per row.
// A per-lane layout: K = base..base+7 and 16+base..16+base+7, base=(lane>=16)*8
// B per-lane layout: same as A (K = base..base+7 and 16+base..16+base+7)
// ---------------------------------------------------------------------------
__global__ void gather_wmma_bce_kernel(const _Float16* __restrict__ fnorm,
                                       const _Float16* __restrict__ pnorm,
                                       const int* __restrict__ sel,
                                       float* __restrict__ out, int bs) {
    int wave = (blockIdx.x * blockDim.x + threadIdx.x) >> 5;
    int lane = threadIdx.x & 31;
    if (wave >= bs) return;  // uniform per wave; EXEC stays all-ones

    const int*       s = sel + wave * 32;
    const _Float16*  f = fnorm + (size_t)wave * DIM;
    const int m    = lane & 15;
    const int base = (lane >= 16) ? 8  : 0;   // A-matrix K sub-base
    const int boff = (lane >= 16) ? 16 : 0;   // unused (B is loaded with base)

    // B vectors for the 4 K-slices (columns all equal f_b)
    v16h bv[4];
    #pragma unroll
    for (int t = 0; t < 4; t++)
        bv[t] = __builtin_shufflevector(*reinterpret_cast<const v8h*>(f + t * 32 + base),
                                        *reinterpret_cast<const v8h*>(f + t * 32 + 16 + base),
                                        0, 1, 2, 3, 4, 5, 6, 7, 8, 9, 10, 11, 12, 13, 14, 15);

    const _Float16* p0 = pnorm + (size_t)s[m]      * DIM;  // slots 0..15
    const _Float16* p1 = pnorm + (size_t)s[16 + m] * DIM;  // slots 16..31

    v8f acc0 = {};
    v8f acc1 = {};
    #pragma unroll
    for (int t = 0; t < 4; t++) {
        const int kk = t * 32;
        v8h lo0 = *reinterpret_cast<const v8h*>(p0 + kk + base);
        v8h hi0 = *reinterpret_cast<const v8h*>(p0 + kk + 16 + base);
        v8h lo1 = *reinterpret_cast<const v8h*>(p1 + kk + base);
        v8h hi1 = *reinterpret_cast<const v8h*>(p1 + kk + 16 + base);
        v16h a0 = __builtin_shufflevector(lo0, hi0, 0, 1, 2, 3, 4, 5, 6, 7,
                                          8, 9, 10, 11, 12, 13, 14, 15);
        v16h a1 = __builtin_shufflevector(lo1, hi1, 0, 1, 2, 3, 4, 5, 6, 7,
                                          8, 9, 10, 11, 12, 13, 14, 15);
        acc0 = __builtin_amdgcn_wmma_f32_16x16x32_f16(false, a0, false, bv[t],
                                                      (short)0, acc0, false, false);
        acc1 = __builtin_amdgcn_wmma_f32_16x16x32_f16(false, a1, false, bv[t],
                                                      (short)0, acc1, false, false);
    }

    // Column N=0 of D: lane 0 holds M=0..7, lane 16 holds M=8..15.
    float local = 0.0f;
    if (lane == 0) {
        #pragma unroll
        for (int r = 0; r < 8; r++) {                 // slots 0..7
            float z = acc0[r] * TEMP_INV;
            float t = (r < KPOS) ? (1.0f / KPOS) : 0.0f;
            local += bce_term(z, t);
        }
        #pragma unroll
        for (int r = 0; r < 8; r++)                   // slots 16..23 (negatives)
            local += bce_term(acc1[r] * TEMP_INV, 0.0f);
    }
    if (lane == 16) {
        #pragma unroll
        for (int r = 0; r < 8; r++)                   // slots 8..15 (negatives)
            local += bce_term(acc0[r] * TEMP_INV, 0.0f);
        // slots 24..31 are padding: skipped
    }
    if (lane == 0 || lane == 16)
        atomicAdd(out, local * (1.0f / ((float)NSLOT * (float)bs)));
}

// ---------------------------------------------------------------------------
// Host-side launcher
// ---------------------------------------------------------------------------
extern "C" void kernel_launch(void* const* d_in, const int* in_sizes, int n_in,
                              void* d_out, int out_size, void* d_ws, size_t ws_size,
                              hipStream_t stream) {
    const float* feat   = (const float*)d_in[0];   // (bs, 128) f32
    const float* prot   = (const float*)d_in[1];   // (C, 128)  f32
    const int*   labels = (const int*)d_in[2];     // (bs, C)   int32
    float*       out    = (float*)d_out;

    const int bs = in_sizes[0] / DIM;              // 4096
    const int C  = in_sizes[1] / DIM;              // 16384

    char* ws = (char*)d_ws;
    _Float16* pnorm = (_Float16*)ws;                                   // C*128 f16 (4 MB)
    _Float16* fnorm = (_Float16*)(ws + (size_t)C * DIM * sizeof(_Float16));
    int*      sel   = (int*)(ws + (size_t)(C + bs) * DIM * sizeof(_Float16));

    zero_out_kernel<<<1, 32, 0, stream>>>(out, out_size);

    // 8 waves per 256-thread block, one wave per row
    normalize_rows_f16<<<C / 8, 256, 0, stream>>>(prot, pnorm, C);
    normalize_rows_f16<<<bs / 8, 256, 0, stream>>>(feat, fnorm, bs);

    // one workgroup per row scans its label row (HBM roofline term)
    select_indices_kernel<<<bs, 256, 0, stream>>>(labels, sel, C);

    // one wave per feature row does the gathered WMMA + BCE
    gather_wmma_bce_kernel<<<bs / 8, 256, 0, stream>>>(fnorm, pnorm, sel, out, bs);
}
